// GIN_35802847380110
// MI455X (gfx1250) — matrix operand order repaired and checked by reading, not compiled, not run on hardware
//
#include <hip/hip_runtime.h>
#include <hip/hip_bf16.h>

// ---------------------------------------------------------------------------
// GIN forward for MI455X (gfx1250): bf16 WMMA GEMMs (f32 accum), hardware
// f32 atomics for segment sums, BN folded into per-column scale/shift.
// ---------------------------------------------------------------------------

#define N_NODES   60000
#define N_EDGES   600000
#define N_FEAT    64
#define HIDDEN    128
#define N_LAYERS  5
#define N_GRAPHS  512
#define N_CLASSES 10
#define BN_EPS    1e-5f

typedef __attribute__((ext_vector_type(8)))  __bf16 v8bf;
typedef __attribute__((ext_vector_type(16))) __bf16 v16bf;
typedef __attribute__((ext_vector_type(8)))  float  v8f;

// -------------------------------- utility ----------------------------------

__global__ void k_zero_f32(float* __restrict__ p, long long n) {
  long long i = (long long)blockIdx.x * blockDim.x + threadIdx.x;
  if (i < n) p[i] = 0.0f;
}

// W [K x N] f32 -> Wt [N x K] bf16 (column-major weights so B-fragment loads
// are two contiguous 16B loads per lane)
__global__ void k_transpose_to_bf16(const float* __restrict__ W,
                                    __bf16* __restrict__ Wt,
                                    int K, int N, long long total) {
  long long i = (long long)blockIdx.x * blockDim.x + threadIdx.x;
  if (i >= total) return;
  int k = (int)(i / N);
  int n = (int)(i % N);
  Wt[(size_t)n * K + k] = (__bf16)W[i];
}

// z[n*F + f] = h[n*hstride + f]   (float4 vectorized)
__global__ void k_copy_strided(const float* __restrict__ H, int hstride,
                               float* __restrict__ Z, int F, long long total4) {
  long long i = (long long)blockIdx.x * blockDim.x + threadIdx.x;
  if (i >= total4) return;
  int nch = F >> 2;
  int n = (int)(i / nch);
  int c = (int)(i % nch) * 4;
  *(float4*)(Z + (size_t)n * F + c) =
      *(const float4*)(H + (size_t)n * hstride + c);
}

// f32 -> bf16 elementwise
__global__ void k_to_bf16(const float* __restrict__ src,
                          __bf16* __restrict__ dst, long long n) {
  long long i = (long long)blockIdx.x * blockDim.x + threadIdx.x;
  if (i < n) dst[i] = (__bf16)src[i];
}

// --------------------------- edge scatter-add ------------------------------
// z[dst] += h[src]  across all edges, 4 features per thread, hardware f32
// atomics (global_atomic_add_f32).
__global__ void k_edge_scatter(const int* __restrict__ src,
                               const int* __restrict__ dst,
                               const float* __restrict__ H, int hstride, int F,
                               float* __restrict__ Z, long long total) {
  long long i = (long long)blockIdx.x * blockDim.x + threadIdx.x;
  if (i >= total) return;
  int nch = F >> 2;
  int e = (int)(i / nch);
  int c = (int)(i % nch) * 4;
  int s = src[e];
  int d = dst[e];
  float4 v = *(const float4*)(H + (size_t)s * hstride + c);
  float* zp = Z + (size_t)d * F + c;
  unsafeAtomicAdd(zp + 0, v.x);
  unsafeAtomicAdd(zp + 1, v.y);
  unsafeAtomicAdd(zp + 2, v.z);
  unsafeAtomicAdd(zp + 3, v.w);
}

// ------------------------------ WMMA GEMM ----------------------------------
// Y[M x 128] = A[M x K](bf16) * Wt[128 x K](bf16, pre-transposed) + bias
// One wave per 16x16 tile, 8 waves/block cover all 128 output columns of a
// 16-row stripe. K consumed 32/step via v_wmma_f32_16x16x32_bf16.
__global__ __launch_bounds__(256)
void k_gemm_bf16_n128(const __bf16* __restrict__ A,
                      const __bf16* __restrict__ Wt,
                      const float* __restrict__ bias,
                      float* __restrict__ Y, int M, int K) {
  const int wave = threadIdx.x >> 5;
  const int lane = threadIdx.x & 31;
  const int half = lane >> 4;     // half-wave selector
  const int lr   = lane & 15;     // row (A) / col (B) within tile
  const int rowBase = blockIdx.x * 16;
  const int colBase = wave * 16;
  if (rowBase >= M) return;

  // A fragment: lane holds row lr; K pattern {0..7,16..23}+8*half per k-step
  const __bf16* arow = A + (size_t)(rowBase + lr) * K + half * 8;
  // B fragment: lane holds col lr; K pattern {0..7,16..23}+8*half per k-step
  const __bf16* bcol = Wt + (size_t)(colBase + lr) * K + half * 8;

  v8f acc = {};
  for (int k = 0; k < K; k += 32) {
    v8bf alo = *(const v8bf*)(arow + k);        // K = k + 8*half + 0..7
    v8bf ahi = *(const v8bf*)(arow + k + 16);   // K = k + 16 + 8*half + 0..7
    v16bf a = __builtin_shufflevector(alo, ahi,
        0, 1, 2, 3, 4, 5, 6, 7, 8, 9, 10, 11, 12, 13, 14, 15);
    v8bf blo = *(const v8bf*)(bcol + k);        // K = k + 8*half + 0..7
    v8bf bhi = *(const v8bf*)(bcol + k + 16);   // K = k + 16 + 8*half + 0..7
    v16bf b = __builtin_shufflevector(blo, bhi,
        0, 1, 2, 3, 4, 5, 6, 7, 8, 9, 10, 11, 12, 13, 14, 15);
    acc = __builtin_amdgcn_wmma_f32_16x16x32_bf16(
        /*neg_a=*/false, a, /*neg_b=*/false, b,
        /*c_mod=*/(short)0, acc, /*reuse_a=*/false, /*reuse_b=*/false);
  }

  const int col = colBase + lr;
  const float bv = bias[col];
#pragma unroll
  for (int i = 0; i < 8; ++i) {
    // C/D layout: VGPR i -> row (i + 8*half), lanes 0..15 -> cols 0..15
    Y[(size_t)(rowBase + i + 8 * half) * 128 + col] = acc[i] + bv;
  }
}

// --------------------------- BatchNorm pieces ------------------------------
// Per-column sum/sumsq partials; 128 threads read a full row each iteration
// (perfectly coalesced), 512 rows per block, f32 atomics for the tails.
__global__ __launch_bounds__(128)
void k_colstats(const float* __restrict__ Y, float* __restrict__ sums,
                float* __restrict__ sumsq, int M) {
  int c  = threadIdx.x;
  int r0 = blockIdx.x * 512;
  int r1 = min(r0 + 512, M);
  float s = 0.0f, q = 0.0f;
  for (int r = r0; r < r1; ++r) {
    float v = Y[(size_t)r * 128 + c];
    s += v;
    q = fmaf(v, v, q);
  }
  unsafeAtomicAdd(&sums[c], s);
  unsafeAtomicAdd(&sumsq[c], q);
}

// Fold BN into per-column scale/shift: y*scale + shift == (y-m)*rsqrt(v+eps)*g+b
__global__ void k_bn_finalize(const float* __restrict__ sums,
                              const float* __restrict__ sumsq,
                              const float* __restrict__ g,
                              const float* __restrict__ be,
                              float* __restrict__ scale,
                              float* __restrict__ shift, float invM) {
  int c = threadIdx.x;
  float m   = sums[c] * invM;
  float var = sumsq[c] * invM - m * m;
  float sc  = g[c] * rsqrtf(var + BN_EPS);
  scale[c] = sc;
  shift[c] = be[c] - m * sc;
}

// h = relu(y*scale + shift); optional bf16 output (next GEMM) and/or strided
// f32 output (JK concat slot, also next layer's input).
__global__ void k_bn_relu(const float* __restrict__ Y,
                          const float* __restrict__ scale,
                          const float* __restrict__ shift,
                          __bf16* __restrict__ out_bf,
                          float* __restrict__ out_f32, int f32stride,
                          long long total) {
  long long i = (long long)blockIdx.x * blockDim.x + threadIdx.x;
  if (i >= total) return;
  int r = (int)(i >> 7);
  int c = (int)(i & 127);
  float v = fmaxf(fmaf(Y[i], scale[c], shift[c]), 0.0f);
  if (out_bf)  out_bf[i] = (__bf16)v;
  if (out_f32) out_f32[(size_t)r * f32stride + c] = v;
}

// ------------------------------- pooling -----------------------------------
__global__ void k_pool_scatter(const float* __restrict__ xs,
                               const int* __restrict__ batch,
                               float* __restrict__ pooled, long long total) {
  long long i = (long long)blockIdx.x * blockDim.x + threadIdx.x;
  if (i >= total) return;
  int n  = (int)(i / 160);
  int c  = (int)(i % 160) * 4;
  float4 v = *(const float4*)(xs + (size_t)n * 640 + c);
  float* p = pooled + (size_t)batch[n] * 640 + c;
  unsafeAtomicAdd(p + 0, v.x);
  unsafeAtomicAdd(p + 1, v.y);
  unsafeAtomicAdd(p + 2, v.z);
  unsafeAtomicAdd(p + 3, v.w);
}

// ------------------------------ final head --------------------------------
__global__ __launch_bounds__(128)
void k_head1(const float* __restrict__ pooled, const float* __restrict__ W,
             const float* __restrict__ b, float* __restrict__ out) {
  int g = blockIdx.x;      // graph
  int j = threadIdx.x;     // output feature
  const float* pr = pooled + (size_t)g * 640;
  float s = b[j];
  for (int k = 0; k < 640; ++k) s = fmaf(pr[k], W[(size_t)k * 128 + j], s);
  out[(size_t)g * 128 + j] = fmaxf(s, 0.0f);
}

__global__ void k_head2(const float* __restrict__ H, const float* __restrict__ W,
                        const float* __restrict__ b, float* __restrict__ out) {
  int g = blockIdx.x;
  int j = threadIdx.x;
  if (j >= N_CLASSES) return;
  const float* hr = H + (size_t)g * 128;
  float s = b[j];
  for (int k = 0; k < 128; ++k) s = fmaf(hr[k], W[(size_t)k * N_CLASSES + j], s);
  out[(size_t)g * N_CLASSES + j] = s;
}

// ------------------------------- launcher ----------------------------------

static inline int nblk(long long total, int bs) {
  return (int)((total + bs - 1) / bs);
}

extern "C" void kernel_launch(void* const* d_in, const int* in_sizes, int n_in,
                              void* d_out, int out_size, void* d_ws, size_t ws_size,
                              hipStream_t stream) {
  (void)in_sizes; (void)n_in; (void)out_size; (void)ws_size;

  // Input order (insertion order, recursive): x, edge_index, batch, params{
  //   convs[l]{lin1.W, lin1.b, g1, be1, lin2.W, lin2.b, g2, be2} x5,
  //   lin1{W,b}, lin2{W,b} }
  const float* x     = (const float*)d_in[0];
  const int*   esrc  = (const int*)d_in[1];            // edge_index[0]
  const int*   edst  = (const int*)d_in[1] + N_EDGES;  // edge_index[1]
  const int*   batch = (const int*)d_in[2];
  auto conv = [&](int l, int t) { return d_in[3 + 8 * l + t]; };
  const float* fin1W = (const float*)d_in[3 + 8 * N_LAYERS + 0];
  const float* fin1b = (const float*)d_in[3 + 8 * N_LAYERS + 1];
  const float* fin2W = (const float*)d_in[3 + 8 * N_LAYERS + 2];
  const float* fin2b = (const float*)d_in[3 + 8 * N_LAYERS + 3];

  // ---- workspace carving (256B aligned) ----
  char* ws = (char*)d_ws;
  size_t off = 0;
  auto carve = [&](size_t bytes) -> char* {
    char* p = ws + off;
    off = (off + bytes + 255) & ~(size_t)255;
    return p;
  };
  __bf16* w1t[N_LAYERS];
  __bf16* w2t[N_LAYERS];
  for (int l = 0; l < N_LAYERS; ++l) {
    int K1 = (l == 0) ? N_FEAT : HIDDEN;
    w1t[l] = (__bf16*)carve((size_t)HIDDEN * K1 * 2);
    w2t[l] = (__bf16*)carve((size_t)HIDDEN * HIDDEN * 2);
  }
  float*  z     = (float*)carve((size_t)N_NODES * HIDDEN * 4);
  __bf16* zb    = (__bf16*)carve((size_t)N_NODES * HIDDEN * 2);
  float*  y     = (float*)carve((size_t)N_NODES * HIDDEN * 4);
  __bf16* h1b   = (__bf16*)carve((size_t)N_NODES * HIDDEN * 2);
  float*  xsAll = (float*)carve((size_t)N_NODES * (N_LAYERS * HIDDEN) * 4);
  float*  sums  = (float*)carve(128 * 4);
  float*  sumsq = (float*)carve(128 * 4);
  float*  scale = (float*)carve(128 * 4);
  float*  shift = (float*)carve(128 * 4);
  float*  pooled = (float*)carve((size_t)N_GRAPHS * (N_LAYERS * HIDDEN) * 4);
  float*  hfin   = (float*)carve((size_t)N_GRAPHS * HIDDEN * 4);

  // ---- weight prep: transpose + bf16 convert ----
  for (int l = 0; l < N_LAYERS; ++l) {
    int K1 = (l == 0) ? N_FEAT : HIDDEN;
    long long t1 = (long long)K1 * HIDDEN;
    k_transpose_to_bf16<<<nblk(t1, 256), 256, 0, stream>>>(
        (const float*)conv(l, 0), w1t[l], K1, HIDDEN, t1);
    long long t2 = (long long)HIDDEN * HIDDEN;
    k_transpose_to_bf16<<<nblk(t2, 256), 256, 0, stream>>>(
        (const float*)conv(l, 4), w2t[l], HIDDEN, HIDDEN, t2);
  }

  const float invM = 1.0f / (float)N_NODES;
  const int statBlocks = nblk(N_NODES, 512);

  for (int l = 0; l < N_LAYERS; ++l) {
    const int F = (l == 0) ? N_FEAT : HIDDEN;
    const float* hin = (l == 0) ? x : (xsAll + (size_t)(l - 1) * HIDDEN);
    const int hstride = (l == 0) ? N_FEAT : (N_LAYERS * HIDDEN);

    // z = h  (self term of GIN, eps = 0)
    long long cp4 = (long long)N_NODES * (F / 4);
    k_copy_strided<<<nblk(cp4, 256), 256, 0, stream>>>(hin, hstride, z, F, cp4);

    // z[dst] += h[src] over edges
    long long et = (long long)N_EDGES * (F / 4);
    k_edge_scatter<<<nblk(et, 256), 256, 0, stream>>>(esrc, edst, hin, hstride,
                                                      F, z, et);

    // bf16 copy of z for WMMA
    long long zt = (long long)N_NODES * F;
    k_to_bf16<<<nblk(zt, 256), 256, 0, stream>>>(z, zb, zt);

    // lin1 + bias  -> y
    k_gemm_bf16_n128<<<N_NODES / 16, 256, 0, stream>>>(zb, w1t[l],
        (const float*)conv(l, 1), y, N_NODES, F);

    // BN1 stats + fold, relu -> h1 (bf16 only; feeds lin2 directly)
    k_zero_f32<<<1, 256, 0, stream>>>(sums, 256);  // sums+sumsq contiguous
    k_colstats<<<statBlocks, 128, 0, stream>>>(y, sums, sumsq, N_NODES);
    k_bn_finalize<<<1, 128, 0, stream>>>(sums, sumsq, (const float*)conv(l, 2),
                                         (const float*)conv(l, 3), scale, shift,
                                         invM);
    long long ht = (long long)N_NODES * HIDDEN;
    k_bn_relu<<<nblk(ht, 256), 256, 0, stream>>>(y, scale, shift, h1b,
                                                 (float*)nullptr, 0, ht);

    // lin2 + bias -> y
    k_gemm_bf16_n128<<<N_NODES / 16, 256, 0, stream>>>(h1b, w2t[l],
        (const float*)conv(l, 5), y, N_NODES, HIDDEN);

    // BN2 + relu -> xsAll[:, l*128 : (l+1)*128]  (f32; next layer's input)
    k_zero_f32<<<1, 256, 0, stream>>>(sums, 256);
    k_colstats<<<statBlocks, 128, 0, stream>>>(y, sums, sumsq, N_NODES);
    k_bn_finalize<<<1, 128, 0, stream>>>(sums, sumsq, (const float*)conv(l, 6),
                                         (const float*)conv(l, 7), scale, shift,
                                         invM);
    k_bn_relu<<<nblk(ht, 256), 256, 0, stream>>>(y, scale, shift,
        (__bf16*)nullptr, xsAll + (size_t)l * HIDDEN, N_LAYERS * HIDDEN, ht);
  }

  // ---- global sum pool over sorted batch ids ----
  long long pn = (long long)N_GRAPHS * (N_LAYERS * HIDDEN);
  k_zero_f32<<<nblk(pn, 256), 256, 0, stream>>>(pooled, pn);
  long long pt = (long long)N_NODES * (N_LAYERS * HIDDEN / 4);
  k_pool_scatter<<<nblk(pt, 256), 256, 0, stream>>>(xsAll, batch, pooled, pt);

  // ---- head ----
  k_head1<<<N_GRAPHS, 128, 0, stream>>>(pooled, fin1W, fin1b, hfin);
  k_head2<<<N_GRAPHS, 16, 0, stream>>>(hfin, fin2W, fin2b, (float*)d_out);
}
